// MSDeformableAttention3D_29566554866346
// MI455X (gfx1250) — compile-verified, matrix operand order checked by reading, not run on hardware
//
#include <hip/hip_runtime.h>

// ---------------- problem constants (match reference) ----------------
#define EDIM 256
#define NHEAD 8
#define NLVL 4
#define NPT 8
#define BSZ 2
#define NQRY 10000
#define LENV 13294            // 100*100 + 50*50 + 25*25 + 13*13
#define DHEAD 32              // EDIM / NHEAD

typedef __attribute__((ext_vector_type(16))) _Float16 v16h;
typedef __attribute__((ext_vector_type(8)))  float    v8f;
typedef __attribute__((ext_vector_type(4)))  float    f32x4;

// Fragment layouts (CDNA5 ISA 7.12.2, 16-bit operands, wave32):
//   A-frag for row-tile tm, k-step ks: lane&15 = M-row, kg=(lane>>4)*8,
//     halfs[0..7] = K(ks*32+kg+0..7), halfs[8..15] = K(ks*32+kg+16..23)
//   B-frag symmetric with N-column in lane&15.
// Packed storage: frag[(tile*8 + ks)*32 + lane] : v16h   (32B per lane)

// ---------------------------------------------------------------------
// Pack A (fp32 [M,256], optional +A2) into fp16 A-fragments.
// One block (256 thr) per row-tile tm; thread = (ks, lane).
// ---------------------------------------------------------------------
template <bool ADD2>
__global__ __launch_bounds__(256)
void pack_a_kernel(const float* __restrict__ A,
                   const float* __restrict__ A2,
                   _Float16* __restrict__ Apack,
                   int M) {
  const int tm   = blockIdx.x;
  const int ks   = threadIdx.x >> 5;
  const int lane = threadIdx.x & 31;

  int row = tm * 16 + (lane & 15);
  row = row < M ? row : M - 1;                   // branchless tail clamp
  const int kg = (lane >> 4) * 8;

  const float* p = A + (size_t)row * EDIM + ks * 32 + kg;
  f32x4 p0 = *(const f32x4*)(p);
  f32x4 p1 = *(const f32x4*)(p + 4);
  f32x4 q0 = *(const f32x4*)(p + 16);
  f32x4 q1 = *(const f32x4*)(p + 20);
  if constexpr (ADD2) {
    const float* p2 = A2 + (size_t)row * EDIM + ks * 32 + kg;
    p0 += *(const f32x4*)(p2);
    p1 += *(const f32x4*)(p2 + 4);
    q0 += *(const f32x4*)(p2 + 16);
    q1 += *(const f32x4*)(p2 + 20);
  }
  v16h a;
#pragma unroll
  for (int j = 0; j < 4; ++j) {
    a[j]      = (_Float16)p0[j];
    a[j + 4]  = (_Float16)p1[j];
    a[j + 8]  = (_Float16)q0[j];
    a[j + 12] = (_Float16)q1[j];
  }
  ((v16h*)Apack)[((size_t)tm * 8 + ks) * 32 + lane] = a;
}

// ---------------------------------------------------------------------
// Pack W (fp32 [256,N]) into fp16 B-fragments. One block per column-tile.
// ---------------------------------------------------------------------
template <int N>
__global__ __launch_bounds__(256)
void pack_b_kernel(const float* __restrict__ W,
                   _Float16* __restrict__ Bpack) {
  const int tn   = blockIdx.x;
  const int ks   = threadIdx.x >> 5;
  const int lane = threadIdx.x & 31;
  const int col  = tn * 16 + (lane & 15);
  const int kg   = (lane >> 4) * 8;

  const float* Wp = W + (size_t)(ks * 32 + kg) * N + col;
  v16h b;
#pragma unroll
  for (int j = 0; j < 8; ++j) {
    b[j]     = (_Float16)Wp[j * N];
    b[j + 8] = (_Float16)Wp[(j + 16) * N];
  }
  ((v16h*)Bpack)[((size_t)tn * 8 + ks) * 32 + lane] = b;
}

// ---------------------------------------------------------------------
// GEMM on pre-packed fp16 fragments: C[M,N] = A @ W + bias.
// One 16x16 tile per wave32; inner loop = 2xb128(A) + 2xb128(B) + 1 wmma.
// Two accumulators break the D->C RAW chain on the XDL pipe.
// ---------------------------------------------------------------------
template <int N>
__global__ __launch_bounds__(256)
void wmma_gemm_kernel(const _Float16* __restrict__ Apack,
                      const _Float16* __restrict__ Bpack,
                      const float* __restrict__ bias,
                      float* __restrict__ C,
                      int M) {
  const int lane = threadIdx.x & 31;
  const int wave = threadIdx.x >> 5;
  constexpr int tilesN = N >> 4;

  const int tile = blockIdx.x * 8 + wave;        // wave-uniform
  const int tm = tile / tilesN;
  const int tn = tile - tm * tilesN;
  if (tm * 16 >= M) return;                      // whole wave exits together

  const v16h* Af = (const v16h*)Apack + (size_t)tm * 8 * 32 + lane;
  const v16h* Bf = (const v16h*)Bpack + (size_t)tn * 8 * 32 + lane;

  v8f acc0 = {}, acc1 = {};
#pragma unroll
  for (int ks = 0; ks < 8; ks += 2) {
    v16h a0 = Af[(ks + 0) * 32];
    v16h b0 = Bf[(ks + 0) * 32];
    v16h a1 = Af[(ks + 1) * 32];
    v16h b1 = Bf[(ks + 1) * 32];
    acc0 = __builtin_amdgcn_wmma_f32_16x16x32_f16(false, a0, false, b0,
                                                  (short)0, acc0, false, false);
    acc1 = __builtin_amdgcn_wmma_f32_16x16x32_f16(false, a1, false, b1,
                                                  (short)0, acc1, false, false);
  }
  v8f acc = acc0 + acc1;

  const int  col   = tn * 16 + (lane & 15);
  const float bv   = bias[col];
  const int  rbase = tm * 16 + ((lane >> 4) << 3);
#pragma unroll
  for (int g = 0; g < 8; ++g) {
    const int r = rbase + g;
    if (r < M) C[(size_t)r * N + col] = acc[g] + bv;
  }
}

// ---------------------------------------------------------------------
// Softmax + bilinear sampling. One wave32 per (b, query, head).
//   lane = channel c in d=32  (corner fetch = 128B coalesced line, L2-hot)
//   lane also indexes the 32 (level,point) logits for the wave softmax.
// v layout: [b][pos][h*32+c]  (straight GEMM output — contiguous per head)
// ---------------------------------------------------------------------
__global__ __launch_bounds__(256)
void deform_sample_kernel(const float* __restrict__ v,
                          const float* __restrict__ off,
                          const float* __restrict__ attn,
                          const float* __restrict__ refp,
                          const int*   __restrict__ shapes,
                          float* __restrict__ mid) {
  const int lane = threadIdx.x & 31;
  const int wave = threadIdx.x >> 5;
  const long gw  = (long)blockIdx.x * 8 + wave;
  const long total = (long)BSZ * NQRY * NHEAD;
  if (gw >= total) return;

  const int  h  = (int)(gw % NHEAD);
  const long bq = gw / NHEAD;                    // b*NQRY + qi
  const int  b  = (int)(bq / NQRY);

  // ---- softmax over 32 logits (lane-parallel, wave32 shuffles) ----
  const float logit = attn[(size_t)bq * (NHEAD * NLVL * NPT) + h * 32 + lane];
  float mx = logit;
#pragma unroll
  for (int s = 16; s; s >>= 1) mx = fmaxf(mx, __shfl_xor(mx, s, 32));
  const float e = __expf(logit - mx);
  float ssum = e;
#pragma unroll
  for (int s = 16; s; s >>= 1) ssum += __shfl_xor(ssum, s, 32);
  const float wgt = e / ssum;

  // ---- preload this lane's own (off_x, off_y); broadcast per point ----
  const float* offp = off + (size_t)bq * (NHEAD * NLVL * NPT * 2) + (size_t)(h * 32) * 2;
  const float ox_l = offp[lane * 2 + 0];
  const float oy_l = offp[lane * 2 + 1];

  // ---- hoist loop-invariant per-level state ----
  int   iH[NLVL], iW[NLVL], lstart[NLVL];
  float prx[NLVL], pry[NLVL];
  int run = 0;
#pragma unroll
  for (int l = 0; l < NLVL; ++l) {
    iH[l] = shapes[l * 2 + 0];
    iW[l] = shapes[l * 2 + 1];
    lstart[l] = run;
    run += iH[l] * iW[l];
    prx[l] = refp[((size_t)bq * NLVL + l) * 2 + 0];
    pry[l] = refp[((size_t)bq * NLVL + l) * 2 + 1];
  }

  const float* vb = v + (size_t)b * LENV * EDIM + h * DHEAD + lane;

  float acc = 0.0f;
#pragma unroll 4
  for (int lp = 0; lp < NLVL * NPT; ++lp) {
    const int l   = lp >> 3;                     // P = 8
    const int iHs = iH[l], iWs = iW[l];

    const float ox = __shfl(ox_l, lp, 32);
    const float oy = __shfl(oy_l, lp, 32);
    const float wl = __shfl(wgt,  lp, 32);

    // grid_sample(align_corners=False): x = loc_x*W - 0.5 = rx*W + ox - 0.5
    const float x = prx[l] * (float)iWs + ox - 0.5f;
    const float y = pry[l] * (float)iHs + oy - 0.5f;

    const float fx0 = floorf(x), fy0 = floorf(y);
    const int   x0 = (int)fx0,  y0 = (int)fy0;
    const int   x1 = x0 + 1,    y1 = y0 + 1;
    const float wx1 = x - fx0,  wy1 = y - fy0;
    const float wx0 = 1.0f - wx1, wy0 = 1.0f - wy1;

    const int base = lstart[l];
    float c = 0.0f;
#pragma unroll
    for (int k = 0; k < 4; ++k) {
      const int xi = (k & 1) ? x1 : x0;
      const int yi = (k & 2) ? y1 : y0;
      const float w = ((k & 1) ? wx1 : wx0) * ((k & 2) ? wy1 : wy0);
      const bool valid = (xi >= 0) & (xi < iWs) & (yi >= 0) & (yi < iHs);
      const int xc = xi < 0 ? 0 : (xi >= iWs ? iWs - 1 : xi);
      const int yc = yi < 0 ? 0 : (yi >= iHs ? iHs - 1 : yi);
      const float val = vb[(size_t)(base + yc * iWs + xc) * EDIM];  // 128B/wave, L2-resident
      c += val * (valid ? w : 0.0f);
    }
    acc += wl * c;
  }

  mid[(size_t)bq * EDIM + h * DHEAD + lane] = acc;
}

// ---------------------------------------------------------------------
extern "C" void kernel_launch(void* const* d_in, const int* in_sizes, int n_in,
                              void* d_out, int out_size, void* d_ws, size_t ws_size,
                              hipStream_t stream) {
  const float* query     = (const float*)d_in[0];
  const float* value     = (const float*)d_in[1];
  const float* query_pos = (const float*)d_in[2];
  const float* refp      = (const float*)d_in[3];
  const int*   shapes    = (const int*)  d_in[4];
  const float* W_off     = (const float*)d_in[5];
  const float* b_off     = (const float*)d_in[6];
  const float* W_attn    = (const float*)d_in[7];
  const float* b_attn    = (const float*)d_in[8];
  const float* W_val     = (const float*)d_in[9];
  const float* b_val     = (const float*)d_in[10];
  const float* W_out     = (const float*)d_in[11];
  const float* b_out     = (const float*)d_in[12];
  float* out = (float*)d_out;

  const int Mv = BSZ * LENV;                     // 26588
  const int Mq = BSZ * NQRY;                     // 20000
  const int tmV = (Mv + 15) / 16;                // 1662
  const int tmQ = (Mq + 15) / 16;                // 1250

  // ---- workspace layout: fp32 intermediates, then fp16 fragments ----
  float* v    = (float*)d_ws;                                      // [Mv,256]
  float* off  = v    + (size_t)Mv * EDIM;                          // [Mq,512]
  float* attn = off  + (size_t)Mq * 512;                           // [Mq,256]
  float* mid  = attn + (size_t)Mq * 256;                           // [Mq,256]
  _Float16* hbase   = (_Float16*)(mid + (size_t)Mq * 256);         // 32B-aligned
  _Float16* Apk_v   = hbase;                                       // tmV*8*32*16
  _Float16* Apk_q   = Apk_v   + (size_t)tmV * 4096;
  _Float16* Apk_mid = Apk_q   + (size_t)tmQ * 4096;
  _Float16* Bpk_val = Apk_mid + (size_t)tmQ * 4096;                // 16*4096
  _Float16* Bpk_off = Bpk_val + (size_t)16 * 4096;                 // 32*4096
  _Float16* Bpk_att = Bpk_off + (size_t)32 * 4096;
  _Float16* Bpk_out = Bpk_att + (size_t)16 * 4096;

  auto gemm_blocks = [](int M, int N) {
    const int tiles = ((M + 15) / 16) * (N / 16);
    return (tiles + 7) / 8;                      // 8 wave-tiles / 256-thr block
  };

  // ---- pack weights (tiny) and activations ----
  pack_b_kernel<256><<<16, 256, 0, stream>>>(W_val,  Bpk_val);
  pack_b_kernel<512><<<32, 256, 0, stream>>>(W_off,  Bpk_off);
  pack_b_kernel<256><<<16, 256, 0, stream>>>(W_attn, Bpk_att);
  pack_b_kernel<256><<<16, 256, 0, stream>>>(W_out,  Bpk_out);
  pack_a_kernel<false><<<tmV, 256, 0, stream>>>(value, nullptr, Apk_v, Mv);
  pack_a_kernel<true ><<<tmQ, 256, 0, stream>>>(query, query_pos, Apk_q, Mq); // shared by 2 GEMMs

  // 1) v = value @ W_val + b_val
  wmma_gemm_kernel<256><<<gemm_blocks(Mv, 256), 256, 0, stream>>>(
      Apk_v, Bpk_val, b_val, v, Mv);
  // 2) off = (q+qp) @ W_off + b_off
  wmma_gemm_kernel<512><<<gemm_blocks(Mq, 512), 256, 0, stream>>>(
      Apk_q, Bpk_off, b_off, off, Mq);
  // 3) attn logits = (q+qp) @ W_attn + b_attn
  wmma_gemm_kernel<256><<<gemm_blocks(Mq, 256), 256, 0, stream>>>(
      Apk_q, Bpk_att, b_attn, attn, Mq);

  // 4) softmax + bilinear gather-accumulate (one wave per (b,q,h))
  {
    const long waves = (long)BSZ * NQRY * NHEAD; // 160000
    deform_sample_kernel<<<(int)((waves + 7) / 8), 256, 0, stream>>>(
        v, off, attn, refp, shapes, mid);
  }

  // 5) out = mid @ W_out + b_out
  pack_a_kernel<false><<<tmQ, 256, 0, stream>>>(mid, nullptr, Apk_mid, Mq);
  wmma_gemm_kernel<256><<<gemm_blocks(Mq, 256), 256, 0, stream>>>(
      Apk_mid, Bpk_out, b_out, out, Mq);
}
